// Mamba_47485158425017
// MI455X (gfx1250) — compile-verified, matrix operand order checked by reading, not run on hardware
//
#include <hip/hip_runtime.h>
#include <hip/hip_bf16.h>
#include <cmath>

// Mamba block forward for MI455X (gfx1250, wave32, WMMA, async LDS copies).
// B=1, L=2048, D_MODEL=1024, D_INNER=2048, D_STATE=16, DT_RANK=64, D_CONV=4.

#define D_MODEL 1024
#define D_INNER 2048
#define D_STATE 16
#define DT_RANK 64
#define D_CONV  4
#define SEQ_L   2048

typedef unsigned int uint;
typedef unsigned long long uptr;
typedef __attribute__((ext_vector_type(16))) __bf16 v16bf;
typedef __attribute__((ext_vector_type(8)))  float  v8f;
typedef __attribute__((__vector_size__(4 * sizeof(int)))) int v4i;

// gfx1250 async global->LDS copy path (ASYNCcnt), guarded so the file still
// compiles (with ds_store staging) if the toolchain lacks the builtin.
#if defined(__gfx1250__) && __has_builtin(__builtin_amdgcn_global_load_async_to_lds_b128)
#define ASYNC_COPY_OK 1
#else
#define ASYNC_COPY_OK 0
#endif

typedef __attribute__((address_space(1))) v4i gv4i;   // global int4
typedef __attribute__((address_space(3))) v4i lv4i;   // LDS int4

__device__ __forceinline__ gv4i* as_gv(const void* p) {
    return (gv4i*)(uptr)p;                // global generic value == AS1 value
}
__device__ __forceinline__ lv4i* as_lv(const void* p) {
    return (lv4i*)(uint)(uptr)p;          // ISA: flat LDS addr[31:0] = LDS offset
}

__device__ __forceinline__ float sigmoidf_(float x) {
    return 1.0f / (1.0f + __expf(-x));
}

// ---------------------------------------------------------------------------
// f32 -> bf16 row-major copy (4 elems/thread, count multiple of 1024).
// ---------------------------------------------------------------------------
__global__ __launch_bounds__(256) void to_bf16_kernel(
    const float* __restrict__ src, __bf16* __restrict__ dst)
{
    int i = (blockIdx.x * 256 + threadIdx.x) * 4;
    #pragma unroll
    for (int j = 0; j < 4; ++j) dst[i + j] = (__bf16)src[i + j];
}

// ---------------------------------------------------------------------------
// Weight prep: W[K][N] (f32) -> Wt[Npad][K] (bf16), rows >= N zero-filled.
// ---------------------------------------------------------------------------
__global__ __launch_bounds__(256) void wt_transpose_kernel(
    const float* __restrict__ W, __bf16* __restrict__ Wt, int K, int N)
{
    int idx = blockIdx.x * 256 + threadIdx.x;   // n * K + k
    int n = idx / K, k = idx - n * K;
    Wt[idx] = (n < N) ? (__bf16)W[(size_t)k * N + n] : (__bf16)0.0f;
}

// ---------------------------------------------------------------------------
// NT GEMM: C[M,N] = A[M,K] * Bt[N,K]^T, bf16 in / f32 accumulate via
// v_wmma_f32_16x16x32_bf16.
//
// Block 256 threads = 8 waves; block tile 128x64; wave tile 32x32 (4 wmma
// per wave per K-step of 32). LDS tiles are stored in WMMA fragment layout
// (ISA 7.12.2): element (row,k) -> lane = row + 16*((k>>3)&1),
// slot = (k&7) + 8*(k>>4); at uint granularity each aligned 16B global chunk
// maps to an aligned 16B LDS chunk, so staging is pure B128 traffic and each
// lane reads its 32B fragment with two ds_load_b128.
//
// Double-buffered: tile k+1 is staged with global_load_async_to_lds_b128
// (ASYNCcnt) while tile k's WMMAs run; s_wait_asynccnt 0 + barrier closes
// the iteration. Falls back to ds_store_b128 staging if no async builtin.
// ---------------------------------------------------------------------------
template <int EPI>   // 0 = plain store, 1 = +bias[n] then softplus
__global__ __launch_bounds__(256) void gemm_wmma_nt(
    const __bf16* __restrict__ A,   // M x K, row-major
    const __bf16* __restrict__ Bt,  // Npad x K, row-major (B transposed)
    const float*  __restrict__ bias,
    float* __restrict__ C,
    int Nreal, int K, int lda, int ldb, int ldc)
{
    __shared__ __align__(32) uint ldsA[2][8 * 32 * 8];   // 2 x 8 A-tiles
    __shared__ __align__(32) uint ldsB[2][4 * 32 * 8];   // 2 x 4 B-tiles

    const int tid  = threadIdx.x;
    const int wave = tid >> 5;
    const int lane = tid & 31;
    const int wrow = wave & 3;      // A tiles 2*wrow, 2*wrow+1
    const int wcol = wave >> 2;     // B tiles 2*wcol, 2*wcol+1
    const int bm = blockIdx.y * 128;
    const int bn = blockIdx.x * 64;

    v8f acc00 = {}, acc01 = {}, acc10 = {}, acc11 = {};

    // --- staging index precompute (uint = bf16-pair granularity) ---
    const int pa  = tid * 8;            // A: 2048 pairs, 8/thread
    const int am  = pa >> 4;            // block-tile row 0..127
    const int akq = pa & 15;            // starting k-pair: 0 or 8
    const int amt = am >> 4, amr = am & 15;
    const int offA0 = (amt * 32 + amr     ) * 8 + ((akq >> 3) ? 4 : 0);
    const int offA1 = (amt * 32 + amr + 16) * 8 + ((akq >> 3) ? 4 : 0);

    const int pb  = tid * 4;            // B: 1024 pairs, 4/thread
    const int bn_ = pb >> 4;            // Bt row 0..63
    const int bkq = pb & 15;            // 0,4,8,12
    const int bnt = bn_ >> 4, bnr = bn_ & 15;
    const int offB = (bnt * 32 + bnr + (((bkq >> 2) & 1) ? 16 : 0)) * 8
                     + ((bkq >> 3) ? 4 : 0);

    const uint* gA = reinterpret_cast<const uint*>(A  + (size_t)(bm + am ) * lda) + akq;
    const uint* gB = reinterpret_cast<const uint*>(Bt + (size_t)(bn + bn_) * ldb) + bkq;

    auto stage = [&](int k0, int buf) {
        const uint* ga = gA + (k0 >> 1);
        const uint* gb = gB + (k0 >> 1);
        uint* da0 = &ldsA[buf][offA0];
        uint* da1 = &ldsA[buf][offA1];
        uint* db  = &ldsB[buf][offB];
#if ASYNC_COPY_OK
        __builtin_amdgcn_global_load_async_to_lds_b128(as_gv(ga),     as_lv(da0), 0, 0);
        __builtin_amdgcn_global_load_async_to_lds_b128(as_gv(ga + 4), as_lv(da1), 0, 0);
        __builtin_amdgcn_global_load_async_to_lds_b128(as_gv(gb),     as_lv(db),  0, 0);
#else
        #pragma unroll
        for (int j = 0; j < 4; ++j) da0[j] = ga[j];
        #pragma unroll
        for (int j = 0; j < 4; ++j) da1[j] = ga[4 + j];
        #pragma unroll
        for (int j = 0; j < 4; ++j) db[j]  = gb[j];
#endif
    };

    const int fA0 = ((2 * wrow    ) * 32 + lane) * 8;
    const int fA1 = ((2 * wrow + 1) * 32 + lane) * 8;
    const int fB0 = ((2 * wcol    ) * 32 + lane) * 8;
    const int fB1 = ((2 * wcol + 1) * 32 + lane) * 8;

    // Prologue: stage tile 0.
    stage(0, 0);
#if ASYNC_COPY_OK
    asm volatile("s_wait_asynccnt 0x0" ::: "memory");
#endif
    __syncthreads();

    for (int k0 = 0; k0 < K; k0 += 32) {
        const int cur = (k0 >> 5) & 1;
        if (k0 + 32 < K) {
            stage(k0 + 32, cur ^ 1);               // overlaps with WMMAs below
            if (k0 + 64 < K)                        // warm L2 two tiles ahead
                __builtin_prefetch(gA + ((k0 + 64) >> 1), 0, 1);
        }

        const v16bf a0 = *reinterpret_cast<const v16bf*>(&ldsA[cur][fA0]);
        const v16bf a1 = *reinterpret_cast<const v16bf*>(&ldsA[cur][fA1]);
        const v16bf b0 = *reinterpret_cast<const v16bf*>(&ldsB[cur][fB0]);
        const v16bf b1 = *reinterpret_cast<const v16bf*>(&ldsB[cur][fB1]);

        // Zig-zag order: each wmma reuses one operand of the previous one.
        acc00 = __builtin_amdgcn_wmma_f32_16x16x32_bf16(false, a0, false, b0, (short)0, acc00, false, false);
        acc10 = __builtin_amdgcn_wmma_f32_16x16x32_bf16(false, a1, false, b0, (short)0, acc10, false, true );
        acc11 = __builtin_amdgcn_wmma_f32_16x16x32_bf16(false, a1, false, b1, (short)0, acc11, true,  false);
        acc01 = __builtin_amdgcn_wmma_f32_16x16x32_bf16(false, a0, false, b1, (short)0, acc01, false, true );

#if ASYNC_COPY_OK
        asm volatile("s_wait_asynccnt 0x0" ::: "memory");
#endif
        __syncthreads();
    }

    // D layout: N = lane&15; VGPR r holds M = r (lanes 0-15) / r+8 (16-31).
    const int cn = lane & 15;
    const int ro = (lane < 16) ? 0 : 8;
    const int n0 = bn + (2 * wcol) * 16 + cn;
    const int n1 = n0 + 16;
    float bias0 = 0.0f, bias1 = 0.0f;
    if (EPI == 1) { bias0 = bias[n0]; bias1 = bias[n1]; }
    #pragma unroll
    for (int r = 0; r < 8; ++r) {
        int m0 = bm + (2 * wrow) * 16 + r + ro;
        int m1 = m0 + 16;
        float v00 = acc00[r], v01 = acc01[r], v10 = acc10[r], v11 = acc11[r];
        if (EPI == 1) {
            v00 += bias0; v00 = (v00 > 20.0f) ? v00 : log1pf(__expf(v00));
            v10 += bias0; v10 = (v10 > 20.0f) ? v10 : log1pf(__expf(v10));
            v01 += bias1; v01 = (v01 > 20.0f) ? v01 : log1pf(__expf(v01));
            v11 += bias1; v11 = (v11 > 20.0f) ? v11 : log1pf(__expf(v11));
        }
        if (n0 < Nreal) {
            C[(size_t)m0 * ldc + n0] = v00;
            C[(size_t)m1 * ldc + n0] = v10;
        }
        if (n1 < Nreal) {
            C[(size_t)m0 * ldc + n1] = v01;
            C[(size_t)m1 * ldc + n1] = v11;
        }
    }
}

// ---------------------------------------------------------------------------
// Depthwise causal conv1d (D_CONV=4) + bias + SiLU; writes f32 (scan) and
// bf16 (A-matrix of x_proj GEMM).
// ---------------------------------------------------------------------------
__global__ __launch_bounds__(256) void conv_silu_kernel(
    const float* __restrict__ xr, const float* __restrict__ w,
    const float* __restrict__ b, float* __restrict__ out_f,
    __bf16* __restrict__ out_b)
{
    int idx = blockIdx.x * 256 + threadIdx.x;
    int d = idx & (D_INNER - 1);
    int l = idx >> 11;
    float acc = b[d];
    #pragma unroll
    for (int j = 0; j < D_CONV; ++j) {
        int li = l - (D_CONV - 1) + j;
        if (li >= 0)
            acc += xr[(size_t)li * (2 * D_INNER) + d] * w[d * D_CONV + j];
    }
    float v = acc * sigmoidf_(acc);
    out_f[idx] = v;
    out_b[idx] = (__bf16)v;
}

// ---------------------------------------------------------------------------
// Selective scan: one thread per channel; 16 recurrence states in VGPRs,
// sequential over L. Per-step B/C broadcast through LDS.
// ---------------------------------------------------------------------------
__global__ __launch_bounds__(256) void scan_kernel(
    const float* __restrict__ u,      // (L, D_INNER)
    const float* __restrict__ delta,  // (L, D_INNER)
    const float* __restrict__ xdbl,   // (L, 96): [64,80)=B [80,96)=C
    const float* __restrict__ A_log,  // (D_INNER, 16)
    const float* __restrict__ Dv,     // (D_INNER)
    float* __restrict__ y)            // (L, D_INNER)
{
    __shared__ float sBC[32];
    const int d = blockIdx.x * 256 + threadIdx.x;

    float Arow[D_STATE];
    #pragma unroll
    for (int n = 0; n < D_STATE; ++n)
        Arow[n] = -__expf(A_log[(size_t)d * D_STATE + n]);
    const float Dd = Dv[d];

    float st[D_STATE];
    #pragma unroll
    for (int n = 0; n < D_STATE; ++n) st[n] = 0.0f;

    for (int t = 0; t < SEQ_L; ++t) {
        if (threadIdx.x < 32)
            sBC[threadIdx.x] = xdbl[(size_t)t * 96 + DT_RANK + threadIdx.x];
        __syncthreads();
        const float dt = delta[(size_t)t * D_INNER + d];
        const float ut = u[(size_t)t * D_INNER + d];
        float acc = 0.0f;
        #pragma unroll
        for (int n = 0; n < D_STATE; ++n) {
            float dA = __expf(dt * Arow[n]);
            st[n] = dA * st[n] + (dt * sBC[n]) * ut;
            acc += st[n] * sBC[16 + n];
        }
        y[(size_t)t * D_INNER + d] = acc + ut * Dd;
        __syncthreads();
    }
}

// ---------------------------------------------------------------------------
// Gating: out_bf16 = y_scan * silu(res); feeds out_proj GEMM directly.
// ---------------------------------------------------------------------------
__global__ __launch_bounds__(256) void gate_kernel(
    const float* __restrict__ ys, const float* __restrict__ xr,
    __bf16* __restrict__ out)
{
    int idx = blockIdx.x * 256 + threadIdx.x;
    int d = idx & (D_INNER - 1);
    int l = idx >> 11;
    float r = xr[(size_t)l * (2 * D_INNER) + D_INNER + d];
    out[idx] = (__bf16)(ys[idx] * r * sigmoidf_(r));
}

// ---------------------------------------------------------------------------
extern "C" void kernel_launch(void* const* d_in, const int* in_sizes, int n_in,
                              void* d_out, int out_size, void* d_ws, size_t ws_size,
                              hipStream_t stream)
{
    const float* hs         = (const float*)d_in[0];   // (2048, 1024)
    const float* in_proj_w  = (const float*)d_in[1];   // (1024, 4096)
    const float* conv_w     = (const float*)d_in[2];   // (2048, 1, 4)
    const float* conv_b     = (const float*)d_in[3];   // (2048)
    const float* x_proj_w   = (const float*)d_in[4];   // (2048, 96)
    const float* dt_proj_w  = (const float*)d_in[5];   // (64, 2048)
    const float* dt_proj_b  = (const float*)d_in[6];   // (2048)
    const float* A_log      = (const float*)d_in[7];   // (2048, 16)
    const float* Dv         = (const float*)d_in[8];   // (2048)
    const float* out_proj_w = (const float*)d_in[9];   // (2048, 1024)
    float* out = (float*)d_out;                        // (2048, 1024)

    // ---- workspace layout ----
    float* fws       = (float*)d_ws;
    float* x_and_res = fws;                                      // 2048*4096
    float* xconv     = x_and_res + (size_t)SEQ_L * 2 * D_INNER;  // 2048*2048
    float* x_dbl     = xconv + (size_t)SEQ_L * D_INNER;          // 2048*96
    float* delta     = x_dbl + (size_t)SEQ_L * 96;               // 2048*2048
    float* yscan     = delta + (size_t)SEQ_L * D_INNER;          // 2048*2048
    __bf16* bws      = (__bf16*)(yscan + (size_t)SEQ_L * D_INNER);
    __bf16* hs_bf    = bws;                                      // 2048*1024
    __bf16* inWt     = hs_bf + (size_t)SEQ_L * D_MODEL;          // 4096*1024
    __bf16* xprojWt  = inWt + (size_t)(2 * D_INNER) * D_MODEL;   // 128*2048
    __bf16* dtWt     = xprojWt + (size_t)128 * D_INNER;          // 2048*64
    __bf16* outWt    = dtWt + (size_t)D_INNER * DT_RANK;         // 1024*2048
    __bf16* xconv_bf = outWt + (size_t)D_MODEL * D_INNER;        // 2048*2048
    __bf16* xdbl_bf  = xconv_bf + (size_t)SEQ_L * D_INNER;       // 2048*96
    __bf16* ygate_bf = (__bf16*)delta;  // reuse: delta dead after the scan

    dim3 blk(256);

    // ---- prep: bf16 activations + transposed bf16 weights ----
    to_bf16_kernel<<<(SEQ_L * D_MODEL) / 1024, blk, 0, stream>>>(hs, hs_bf);
    wt_transpose_kernel<<<(4096 * 1024) / 256, blk, 0, stream>>>(in_proj_w, inWt, 1024, 4096);
    wt_transpose_kernel<<<(128 * 2048) / 256, blk, 0, stream>>>(x_proj_w, xprojWt, 2048, 96);
    wt_transpose_kernel<<<(2048 * 64) / 256, blk, 0, stream>>>(dt_proj_w, dtWt, 64, 2048);
    wt_transpose_kernel<<<(1024 * 2048) / 256, blk, 0, stream>>>(out_proj_w, outWt, 2048, 1024);

    // 1) in_proj: (2048x1024)x(1024x4096) -> x_and_res
    gemm_wmma_nt<0><<<dim3(4096 / 64, 2048 / 128), blk, 0, stream>>>(
        hs_bf, inWt, nullptr, x_and_res, 4096, 1024, 1024, 1024, 4096);

    // 2) conv + SiLU -> xconv (f32 + bf16)
    conv_silu_kernel<<<(SEQ_L * D_INNER) / 256, blk, 0, stream>>>(
        x_and_res, conv_w, conv_b, xconv, xconv_bf);

    // 3) x_proj: (2048x2048)x(2048x96) -> x_dbl (Npad=128, masked store)
    gemm_wmma_nt<0><<<dim3(128 / 64, 2048 / 128), blk, 0, stream>>>(
        xconv_bf, xprojWt, nullptr, x_dbl, 96, 2048, 2048, 2048, 96);

    // 3b) bf16 copy of x_dbl (dt columns feed dt_proj GEMM)
    to_bf16_kernel<<<(SEQ_L * 96) / 1024, blk, 0, stream>>>(x_dbl, xdbl_bf);

    // 4) dt_proj + bias + softplus: (2048x64)x(64x2048) -> delta
    gemm_wmma_nt<1><<<dim3(2048 / 64, 2048 / 128), blk, 0, stream>>>(
        xdbl_bf, dtWt, dt_proj_b, delta, 2048, 64, 96, 64, 2048);

    // 5) selective scan -> yscan
    scan_kernel<<<D_INNER / 256, blk, 0, stream>>>(
        xconv, delta, x_dbl, A_log, Dv, yscan);

    // 6) gating -> ygate_bf (reuses delta storage)
    gate_kernel<<<(SEQ_L * D_INNER) / 256, blk, 0, stream>>>(
        yscan, x_and_res, ygate_bf);

    // 7) out_proj: (2048x2048)x(2048x1024) -> out
    gemm_wmma_nt<0><<<dim3(1024 / 64, 2048 / 128), blk, 0, stream>>>(
        ygate_bf, outWt, nullptr, out, 1024, 2048, 2048, 2048, 1024);
}
